// Generator_23235773071432
// MI455X (gfx1250) — matrix operand order repaired and checked by reading, not compiled, not run on hardware
//
#include <hip/hip_runtime.h>

#define NNODES   10000
#define NBATCH   2
#define NT       20000            // NBATCH * NNODES
#define ME       320000           // template edges
#define ETOT     660000           // 2*ME + NT
#define FDIM     64
#define STY      128
#define LAT      128
#define HEADS    4
#define HID      32
#define C01      128              // HEADS*HID
#define K0       192              // FDIM + STY
#define OUTD     67
#define NEGS     0.2f

typedef __attribute__((ext_vector_type(16))) _Float16 v16h;
typedef __attribute__((ext_vector_type(8)))  float    v8f;

__device__ __forceinline__ float leaky(float x) { return x > 0.f ? x : NEGS * x; }

__device__ __forceinline__ unsigned f2o(float f) {
    unsigned u = __float_as_uint(f);
    return (u & 0x80000000u) ? ~u : (u | 0x80000000u);
}
__device__ __forceinline__ float o2f(unsigned k) {
    return (k & 0x80000000u) ? __uint_as_float(k & 0x7FFFFFFFu) : __uint_as_float(~k);
}

__device__ __forceinline__ void edge_sd(int e, const int* __restrict__ ei, int& s, int& d) {
    if (e < ME)            { s = ei[e];            d = ei[ME + e]; }
    else if (e < 2 * ME)   { int i = e - ME; s = ei[i] + NNODES; d = ei[ME + i] + NNODES; }
    else                   { s = d = e - 2 * ME; }
}

__device__ __forceinline__ float wave_sum(float v) {
    #pragma unroll
    for (int m = 16; m > 0; m >>= 1) v += __shfl_xor(v, m, 32);
    return v;
}

// ---------------- style MLP: [2,128]->256->512->128 ------------------------
__global__ void style_mlp_k(const float* __restrict__ z,
                            const float* __restrict__ w1, const float* __restrict__ b1,
                            const float* __restrict__ w2, const float* __restrict__ b2,
                            const float* __restrict__ w3, const float* __restrict__ b3,
                            float* __restrict__ style) {
    __shared__ float zz[NBATCH * LAT];
    __shared__ float s1[NBATCH * 256];
    __shared__ float s2[NBATCH * 512];
    int tid = threadIdx.x;
    zz[tid] = z[tid];                               // 256 threads, 256 elems
    __syncthreads();
    for (int idx = tid; idx < NBATCH * 256; idx += 256) {
        int b = idx >> 8, j = idx & 255;
        float a = b1[j];
        for (int k = 0; k < LAT; ++k) a += zz[b * LAT + k] * w1[k * 256 + j];
        s1[idx] = leaky(a);
    }
    __syncthreads();
    for (int idx = tid; idx < NBATCH * 512; idx += 256) {
        int b = idx >> 9, j = idx & 511;
        float a = b2[j];
        for (int k = 0; k < 256; ++k) a += s1[b * 256 + k] * w2[k * 512 + j];
        s2[idx] = leaky(a);
    }
    __syncthreads();
    for (int idx = tid; idx < NBATCH * STY; idx += 256) {
        int b = idx >> 7, j = idx & 127;
        float a = b3[j];
        for (int k = 0; k < 512; ++k) a += s2[b * 512 + k] * w3[k * STY + j];
        style[idx] = a;
    }
}

// ---------------- H0 = concat(tile(x), repeat(style)) ----------------------
__global__ void build_h0_k(const float* __restrict__ x, const float* __restrict__ style,
                           float* __restrict__ H0) {
    int t = blockIdx.x * blockDim.x + threadIdx.x;
    if (t >= NT * K0) return;
    int n = t / K0, c = t - n * K0;
    H0[t] = (c < FDIM) ? x[(n % NNODES) * FDIM + c]
                       : style[(n / NNODES) * STY + (c - FDIM)];
}

// ---------------- pack W[K,Nw] into f16 WMMA B-fragments -------------------
// layout: Bp[((ct*Ksteps + ks)*32 + lane)*16 + j]
//   lane&15 = n within tile, lane>>4 selects K half; j = 0..15
__global__ void pack_b_k(const float* __restrict__ W, _Float16* __restrict__ Bp,
                         int K, int Nw, int Ntiles) {
    int t = blockIdx.x * blockDim.x + threadIdx.x;
    int Ksteps = K >> 5;
    int total = Ntiles * Ksteps * 32 * 16;
    if (t >= total) return;
    int j    = t & 15;
    int lane = (t >> 4) & 31;
    int rest = t >> 9;
    int ks = rest % Ksteps;
    int ct = rest / Ksteps;
    int n = ct * 16 + (lane & 15);
    int k = ks * 32 + (j & 8) * 2 + (lane >> 4) * 8 + (j & 7);
    Bp[t] = (n < Nw) ? (_Float16)W[(size_t)k * Nw + n] : (_Float16)0.f;
}

// ---------------- WMMA GEMM: C[M,Nw] = A[M,K] * Bpacked (f16 in, f32 acc) --
// one wave per 16x16 tile; K compile-time multiple of 32; M multiple of 16
template <int K>
__global__ void wmma_gemm_k(const float* __restrict__ A, const _Float16* __restrict__ Bp,
                            float* __restrict__ C, int Nw, int Ntiles) {
    constexpr int KS = K / 32;
    int ct = blockIdx.y * blockDim.y + threadIdx.y;
    if (ct >= Ntiles) return;                       // wave-uniform exit
    int lane = threadIdx.x;
    int m = lane & 15, half = lane >> 4;
    int r0 = blockIdx.x * 16;
    v8f acc = {};
    const float* arow = A + (size_t)(r0 + m) * K + half * 8;
    const v16h*  bfr  = (const v16h*)Bp + (size_t)ct * KS * 32 + lane;
    #pragma unroll
    for (int ks = 0; ks < KS; ++ks) {
        const float4* ap = (const float4*)(arow + ks * 32);
        float4 f0 = ap[0], f1 = ap[1], f2 = ap[4], f3 = ap[5];
        v16h a;
        a[0]=(_Float16)f0.x;  a[1]=(_Float16)f0.y;  a[2]=(_Float16)f0.z;  a[3]=(_Float16)f0.w;
        a[4]=(_Float16)f1.x;  a[5]=(_Float16)f1.y;  a[6]=(_Float16)f1.z;  a[7]=(_Float16)f1.w;
        a[8]=(_Float16)f2.x;  a[9]=(_Float16)f2.y;  a[10]=(_Float16)f2.z; a[11]=(_Float16)f2.w;
        a[12]=(_Float16)f3.x; a[13]=(_Float16)f3.y; a[14]=(_Float16)f3.z; a[15]=(_Float16)f3.w;
        v16h b = bfr[ks * 32];
        acc = __builtin_amdgcn_wmma_f32_16x16x32_f16(false, a, false, b,
                                                     (short)0, acc, false, false);
    }
    int col = ct * 16 + m;
    if (col < Nw) {
        #pragma unroll
        for (int g = 0; g < 8; ++g)
            C[(size_t)(r0 + half * 8 + g) * Nw + col] = acc[g];
    }
}

// ---------------- per-node attention logits --------------------------------
__global__ void attn_logits_k(const float* __restrict__ ht,
                              const float* __restrict__ as_, const float* __restrict__ ad_,
                              float* __restrict__ als, float* __restrict__ ald,
                              int H, int D) {
    int t = blockIdx.x * blockDim.x + threadIdx.x;
    if (t >= NT * H) return;
    int n = t / H, h = t - n * H;
    const float* row = ht + (size_t)n * H * D + h * D;
    float ss = 0.f, sd = 0.f;
    for (int i = 0; i < D; ++i) {
        float v = row[i];
        ss += v * as_[h * D + i];
        sd += v * ad_[h * D + i];
    }
    als[t] = ss;
    ald[t] = sd;
}

// ---------------- edge pass 1: segment max ---------------------------------
__global__ void edge_max_k(const int* __restrict__ ei,
                           const float* __restrict__ als, const float* __restrict__ ald,
                           unsigned* __restrict__ emax, int H) {
    int t = blockIdx.x * blockDim.x + threadIdx.x;
    if (t >= ETOT * H) return;
    int e = t / H, h = t - e * H;
    int s, d;
    edge_sd(e, ei, s, d);
    float v = leaky(als[s * H + h] + ald[d * H + h]);
    atomicMax(&emax[d * H + h], f2o(v));
}

// ---------------- edge pass 2: exp/denom/message accumulate ----------------
__global__ void edge_accum_k(const int* __restrict__ ei,
                             const float* __restrict__ als, const float* __restrict__ ald,
                             const unsigned* __restrict__ emax,
                             const float* __restrict__ ht,
                             float* __restrict__ agg, float* __restrict__ denom,
                             int H, int D) {
    int wave = (blockIdx.x * blockDim.x + threadIdx.x) >> 5;
    int lane = threadIdx.x & 31;
    if (wave >= ETOT) return;
    int s, d;
    edge_sd(wave, ei, s, d);
    for (int h = 0; h < H; ++h) {
        float e  = leaky(als[s * H + h] + ald[d * H + h]);
        float mx = o2f(emax[d * H + h]);
        float ex = __expf(e - mx);
        if (lane == 0) atomicAdd(&denom[d * H + h], ex);
        for (int dd = lane; dd < D; dd += 32)
            atomicAdd(&agg[(size_t)d * H * D + h * D + dd],
                      ex * ht[(size_t)s * H * D + h * D + dd]);
    }
}

// ---------------- finalize + LayerNorm + leaky (C = 128, H = 4) ------------
__global__ void finalize_ln_k(const float* __restrict__ agg, const float* __restrict__ denom,
                              const float* __restrict__ bias,
                              const float* __restrict__ lg, const float* __restrict__ lb,
                              float* __restrict__ out) {
    int n = (blockIdx.x * blockDim.x + threadIdx.x) >> 5;
    int lane = threadIdx.x & 31;
    if (n >= NT) return;
    float t[4], s = 0.f, s2 = 0.f;
    #pragma unroll
    for (int j = 0; j < 4; ++j) {
        int c = j * 32 + lane;
        float v = agg[(size_t)n * C01 + c] / (denom[n * HEADS + j] + 1e-16f) + bias[c];
        t[j] = v; s += v; s2 += v * v;
    }
    s  = wave_sum(s);
    s2 = wave_sum(s2);
    float mu  = s * (1.f / C01);
    float var = s2 * (1.f / C01) - mu * mu;
    float inv = rsqrtf(var + 1e-5f);
    #pragma unroll
    for (int j = 0; j < 4; ++j) {
        int c = j * 32 + lane;
        out[(size_t)n * C01 + c] = leaky((t[j] - mu) * inv * lg[c] + lb[c]);
    }
}

// ---------------- finalize layer 2 (heads=1, no LN) ------------------------
__global__ void finalize2_k(const float* __restrict__ agg, const float* __restrict__ denom,
                            const float* __restrict__ bias, float* __restrict__ out) {
    int t = blockIdx.x * blockDim.x + threadIdx.x;
    if (t >= NT * OUTD) return;
    int n = t / OUTD, c = t - n * OUTD;
    out[t] = agg[t] / (denom[n] + 1e-16f) + bias[c];
}

// ---------------- residual + output pack: f_new[NT*64] then p_new[NT*3] ----
__global__ void out_k(const float* __restrict__ Hout,
                      const float* __restrict__ x, const float* __restrict__ pos,
                      float* __restrict__ out) {
    int t = blockIdx.x * blockDim.x + threadIdx.x;
    if (t >= NT * OUTD) return;
    int n = t / OUTD, c = t - n * OUTD;
    int n0 = n % NNODES;
    float v = Hout[t];
    if (c < 3)
        out[(size_t)NT * FDIM + n * 3 + c] = pos[n0 * 3 + c] + v;       // p_new
    else
        out[(size_t)n * FDIM + (c - 3)] = x[n0 * FDIM + (c - 3)] + v;   // f_new
}

extern "C" void kernel_launch(void* const* d_in, const int* in_sizes, int n_in,
                              void* d_out, int out_size, void* d_ws, size_t ws_size,
                              hipStream_t stream) {
    const float* z    = (const float*)d_in[0];
    const float* x    = (const float*)d_in[1];
    const float* pos  = (const float*)d_in[2];
    const int*   ei   = (const int*)  d_in[3];
    const float* w1   = (const float*)d_in[4];
    const float* b1   = (const float*)d_in[5];
    const float* w2   = (const float*)d_in[6];
    const float* b2   = (const float*)d_in[7];
    const float* w3   = (const float*)d_in[8];
    const float* b3   = (const float*)d_in[9];
    const float* gw0  = (const float*)d_in[10];
    const float* ga0s = (const float*)d_in[11];
    const float* ga0d = (const float*)d_in[12];
    const float* gb0  = (const float*)d_in[13];
    const float* gw1  = (const float*)d_in[14];
    const float* ga1s = (const float*)d_in[15];
    const float* ga1d = (const float*)d_in[16];
    const float* gb1  = (const float*)d_in[17];
    const float* gw2  = (const float*)d_in[18];
    const float* ga2s = (const float*)d_in[19];
    const float* ga2d = (const float*)d_in[20];
    const float* gb2  = (const float*)d_in[21];
    const float* ln0g = (const float*)d_in[22];
    const float* ln0b = (const float*)d_in[23];
    const float* ln1g = (const float*)d_in[24];
    const float* ln1b = (const float*)d_in[25];
    float* out = (float*)d_out;

    // ---- workspace layout (256B aligned slabs) ----
    char* base = (char*)d_ws;
    size_t off = 0;
    auto take = [&](size_t bytes) -> char* {
        char* p = base + off;
        off = (off + bytes + 255) & ~(size_t)255;
        return p;
    };
    float*     style = (float*)    take((size_t)NBATCH * STY * 4);
    float*     H0    = (float*)    take((size_t)NT * K0 * 4);
    float*     ht    = (float*)    take((size_t)NT * C01 * 4);   // reused all layers
    float*     als   = (float*)    take((size_t)NT * HEADS * 4);
    float*     ald   = (float*)    take((size_t)NT * HEADS * 4);
    unsigned*  emax  = (unsigned*) take((size_t)NT * HEADS * 4);
    float*     den   = (float*)    take((size_t)NT * HEADS * 4);
    float*     agg   = (float*)    take((size_t)NT * C01 * 4);
    float*     Hcur  = (float*)    take((size_t)NT * C01 * 4);
    float*     Hout  = (float*)    take((size_t)NT * OUTD * 4);
    // packed f16 weight fragments
    const int NT0 = C01 / 16, NT2 = (OUTD + 15) / 16;
    const int P0 = NT0 * (K0 / 32) * 512;        // halfs
    const int P1 = NT0 * (C01 / 32) * 512;
    const int P2 = NT2 * (C01 / 32) * 512;
    _Float16* bp0 = (_Float16*) take((size_t)P0 * 2);
    _Float16* bp1 = (_Float16*) take((size_t)P1 * 2);
    _Float16* bp2 = (_Float16*) take((size_t)P2 * 2);
    (void)ws_size; (void)n_in; (void)in_sizes; (void)out_size;

    dim3 gblk(32, 4);

    // 0. pack weights into WMMA B-fragment layout
    pack_b_k<<<(P0 + 255) / 256, 256, 0, stream>>>(gw0, bp0, K0,  C01,  NT0);
    pack_b_k<<<(P1 + 255) / 256, 256, 0, stream>>>(gw1, bp1, C01, C01,  NT0);
    pack_b_k<<<(P2 + 255) / 256, 256, 0, stream>>>(gw2, bp2, C01, OUTD, NT2);

    // 1. style MLP
    style_mlp_k<<<1, 256, 0, stream>>>(z, w1, b1, w2, b2, w3, b3, style);

    // 2. H0 = [x || style]
    build_h0_k<<<(NT * K0 + 255) / 256, 256, 0, stream>>>(x, style, H0);

    // ---------------- GAT layer 0 ----------------
    wmma_gemm_k<K0><<<dim3(NT / 16, 2), gblk, 0, stream>>>(H0, bp0, ht, C01, NT0);
    attn_logits_k<<<(NT * HEADS + 255) / 256, 256, 0, stream>>>(ht, ga0s, ga0d, als, ald, HEADS, HID);
    hipMemsetAsync(emax, 0, (size_t)NT * HEADS * 4, stream);
    hipMemsetAsync(den,  0, (size_t)NT * HEADS * 4, stream);
    hipMemsetAsync(agg,  0, (size_t)NT * C01 * 4,   stream);
    edge_max_k<<<(ETOT * HEADS + 255) / 256, 256, 0, stream>>>(ei, als, ald, emax, HEADS);
    edge_accum_k<<<(ETOT * 32 + 255) / 256, 256, 0, stream>>>(ei, als, ald, emax, ht, agg, den, HEADS, HID);
    finalize_ln_k<<<(NT * 32 + 255) / 256, 256, 0, stream>>>(agg, den, gb0, ln0g, ln0b, Hcur);

    // ---------------- GAT layer 1 ----------------
    wmma_gemm_k<C01><<<dim3(NT / 16, 2), gblk, 0, stream>>>(Hcur, bp1, ht, C01, NT0);
    attn_logits_k<<<(NT * HEADS + 255) / 256, 256, 0, stream>>>(ht, ga1s, ga1d, als, ald, HEADS, HID);
    hipMemsetAsync(emax, 0, (size_t)NT * HEADS * 4, stream);
    hipMemsetAsync(den,  0, (size_t)NT * HEADS * 4, stream);
    hipMemsetAsync(agg,  0, (size_t)NT * C01 * 4,   stream);
    edge_max_k<<<(ETOT * HEADS + 255) / 256, 256, 0, stream>>>(ei, als, ald, emax, HEADS);
    edge_accum_k<<<(ETOT * 32 + 255) / 256, 256, 0, stream>>>(ei, als, ald, emax, ht, agg, den, HEADS, HID);
    finalize_ln_k<<<(NT * 32 + 255) / 256, 256, 0, stream>>>(agg, den, gb1, ln1g, ln1b, Hcur);

    // ---------------- GAT layer 2 (heads=1, OUT=67) ----------------
    wmma_gemm_k<C01><<<dim3(NT / 16, 2), gblk, 0, stream>>>(Hcur, bp2, ht, OUTD, NT2);
    attn_logits_k<<<(NT + 255) / 256, 256, 0, stream>>>(ht, ga2s, ga2d, als, ald, 1, OUTD);
    hipMemsetAsync(emax, 0, (size_t)NT * 4, stream);
    hipMemsetAsync(den,  0, (size_t)NT * 4, stream);
    hipMemsetAsync(agg,  0, (size_t)NT * OUTD * 4, stream);
    edge_max_k<<<(ETOT + 255) / 256, 256, 0, stream>>>(ei, als, ald, emax, 1);
    edge_accum_k<<<(ETOT * 32 + 255) / 256, 256, 0, stream>>>(ei, als, ald, emax, ht, agg, den, 1, OUTD);
    finalize2_k<<<(NT * OUTD + 255) / 256, 256, 0, stream>>>(agg, den, gb2, Hout);

    // residual + pack outputs
    out_k<<<(NT * OUTD + 255) / 256, 256, 0, stream>>>(Hout, x, pos, out);
}
